// MultiHeadAttention_3143916061137
// MI455X (gfx1250) — matrix operand order repaired and checked by reading, not compiled, not run on hardware
//
#include <hip/hip_runtime.h>
#include <hip/hip_bf16.h>

typedef _Float16 half_t;
typedef __attribute__((ext_vector_type(16))) _Float16 v16h;
typedef __attribute__((ext_vector_type(8)))  _Float16 v8h;
typedef __attribute__((ext_vector_type(4)))  _Float16 v4h;
typedef __attribute__((ext_vector_type(8)))  float    v8f;

#define D_MODEL 1024
#define SEQ     1024
#define BATCH   4
#define NH      16
#define DHEAD   64
#define GBATCH  (BATCH * NH)      // 64 attention batches after row-major .view
#define MROWS   (BATCH * SEQ)     // 4096 rows in every projection GEMM

// ---------------------------------------------------------------------------
// WMMA helpers (CDNA5 v_wmma_f32_16x16x32_f16, wave32)
// ---------------------------------------------------------------------------
__device__ __forceinline__ v8f wmma_f16(v16h a, v16h b, v8f c) {
  // 8 args: (neg_a, A, neg_b, B, c_mod, C, reuse_a, reuse_b)
  return __builtin_amdgcn_wmma_f32_16x16x32_f16(false, a, false, b, (short)0, c,
                                                false, false);
}

// A-matrix 16x32 f16 fragment from row-major [rows][ldk] region.
// ISA layout: lanes0-15 row=lane, halves0-7 K=8*(lane>=16)+j, halves8-15 K+=16.
__device__ __forceinline__ v16h load_frag_a(const half_t* base, int ldk) {
  const int lane = threadIdx.x & 31;
  const int r    = lane & 15;
  const int khi  = (lane >> 4) & 1;
  const half_t* p = base + r * ldk + 8 * khi;
  union { v16h v; v8h h[2]; } u;
  u.h[0] = *(const v8h*)(p);
  u.h[1] = *(const v8h*)(p + 16);
  return u.v;
}

// B-matrix 32x16 f16 fragment from K-major storage: each "row" of [cols][ldk]
// is one output column's contiguous K-run. lane=column; halves0-7 K=8*(lane>=16)+j, halves8-15 K+=16.
__device__ __forceinline__ v16h load_frag_b(const half_t* base, int ldk) {
  const int lane = threadIdx.x & 31;
  const int c    = lane & 15;
  const int ks   = (lane >> 4) & 1;
  const half_t* p = base + c * ldk + 8 * ks;
  union { v16h v; v8h h[2]; } u;
  u.h[0] = *(const v8h*)(p);
  u.h[1] = *(const v8h*)(p + 16);
  return u.v;
}

// ---------------------------------------------------------------------------
// Weight transpose + f32->f16 convert: out[n][k] = (half)in[k][n]
// ---------------------------------------------------------------------------
__global__ __launch_bounds__(256)
void mha_transpose_cvt_kernel(const float* __restrict__ in,
                              half_t* __restrict__ out, int K, int N) {
  __shared__ float tile[32][33];
  const int tx = threadIdx.x & 31;
  const int ty = threadIdx.x >> 5;          // 0..7
  const int n0 = blockIdx.x * 32;
  const int k0 = blockIdx.y * 32;
#pragma unroll
  for (int i = 0; i < 32; i += 8)
    tile[ty + i][tx] = in[(size_t)(k0 + ty + i) * N + (n0 + tx)];
  __syncthreads();
#pragma unroll
  for (int i = 0; i < 32; i += 8)
    out[(size_t)(n0 + ty + i) * K + (k0 + tx)] = (half_t)tile[tx][ty + i];
}

// ---------------------------------------------------------------------------
// GEMM: C[M,N] = A[M,K](f32) * Bt[N,K](f16, pre-transposed) + bias
// Block tile 128x128, 8 waves of 32x64, K-step 32 -> 8 WMMAs/wave/step.
// Software-pipelined: tile k+1 global loads overlap tile k WMMA compute;
// GL2 prefetch issued 4 K-steps ahead.
// ---------------------------------------------------------------------------
template <bool OUT_F16>
__global__ __launch_bounds__(256)
void mha_gemm_bias_kernel(const float* __restrict__ A,
                          const half_t* __restrict__ Bt,
                          const float* __restrict__ bias,
                          half_t* __restrict__ outh,
                          float* __restrict__ outf,
                          int M, int N, int K) {
  constexpr int LDA = 40;   // 32 + 8 pad (halfs); 80B row stride, 16B aligned
  constexpr int LDB = 40;
  __shared__ __align__(16) half_t A_lds[128 * LDA];
  __shared__ __align__(16) half_t B_lds[128 * LDB];

  const int tid    = threadIdx.x;
  const int wid    = tid >> 5;
  const int wave_m = wid >> 1;              // 0..3 -> 32-row strip
  const int wave_n = wid & 1;               // 0..1 -> 64-col strip
  const int mbase  = blockIdx.y * 128;
  const int nbase  = blockIdx.x * 128;

  v8f zero = {};
  v8f acc[2][4];
#pragma unroll
  for (int i = 0; i < 2; ++i)
#pragma unroll
    for (int j = 0; j < 4; ++j) acc[i][j] = zero;

  // Per-thread staging registers for the pipelined global->LDS path
  float4 aReg[4];
  v8h    bReg[2];

  auto load_tiles = [&](int k0) {
#pragma unroll
    for (int i = 0; i < 4; ++i) {
      const int seg = tid + i * 256;        // 1024 float4 segs, 128x32 f32
      const int row = seg >> 3;
      const int kk  = (seg & 7) * 4;
      aReg[i] = *(const float4*)(A + (size_t)(mbase + row) * K + k0 + kk);
    }
#pragma unroll
    for (int i = 0; i < 2; ++i) {
      const int seg  = tid + i * 256;       // 512 v8h segs, 128 cols x 32 K
      const int nrow = seg >> 2;
      const int kk   = (seg & 3) * 8;
      bReg[i] = *(const v8h*)(Bt + (size_t)(nbase + nrow) * K + k0 + kk);
    }
  };
  auto store_tiles = [&]() {
#pragma unroll
    for (int i = 0; i < 4; ++i) {
      const int seg = tid + i * 256;
      const int row = seg >> 3;
      const int kk  = (seg & 7) * 4;
      const float4 f = aReg[i];
      v4h h = {(half_t)f.x, (half_t)f.y, (half_t)f.z, (half_t)f.w};
      *(v4h*)&A_lds[row * LDA + kk] = h;
    }
#pragma unroll
    for (int i = 0; i < 2; ++i) {
      const int seg  = tid + i * 256;
      const int nrow = seg >> 2;
      const int kk   = (seg & 3) * 8;
      *(v8h*)&B_lds[nrow * LDB + kk] = bReg[i];
    }
  };

  load_tiles(0);
  for (int k0 = 0; k0 < K; k0 += 32) {
    __syncthreads();
    store_tiles();
    __syncthreads();
    // Overlap next tile's global loads with this tile's WMMA compute
    if (k0 + 32 < K) {
      load_tiles(k0 + 32);
      if (k0 + 128 < K) {        // warm GL2 ~4 steps ahead (global_prefetch_b8)
        __builtin_prefetch(A + (size_t)(mbase + (tid >> 3)) * K + k0 + 128, 0, 3);
        __builtin_prefetch(Bt + (size_t)(nbase + (tid >> 2)) * K + k0 + 128, 0, 3);
      }
    }

    v16h afrag[2];
#pragma unroll
    for (int mt = 0; mt < 2; ++mt)
      afrag[mt] = load_frag_a(&A_lds[(wave_m * 32 + mt * 16) * LDA], LDA);
#pragma unroll
    for (int nt = 0; nt < 4; ++nt) {
      const v16h bfrag = load_frag_b(&B_lds[(wave_n * 64 + nt * 16) * LDB], LDB);
#pragma unroll
      for (int mt = 0; mt < 2; ++mt)
        acc[mt][nt] = wmma_f16(afrag[mt], bfrag, acc[mt][nt]);
    }
  }

  // Epilogue: C layout -> lane holds col = lane&15, rows r + 8*(lane>=16)
  const int lane = tid & 31;
  const int cmod = lane & 15;
  const int rhal = (lane >> 4) * 8;
#pragma unroll
  for (int mt = 0; mt < 2; ++mt) {
    const int row0 = mbase + wave_m * 32 + mt * 16 + rhal;
#pragma unroll
    for (int nt = 0; nt < 4; ++nt) {
      const int col = nbase + wave_n * 64 + nt * 16 + cmod;
      const float bval = bias[col];
#pragma unroll
      for (int r = 0; r < 8; ++r) {
        const float v = acc[mt][nt][r] + bval;
        const size_t idx = (size_t)(row0 + r) * N + col;
        if (OUT_F16) outh[idx] = (half_t)v;
        else         outf[idx] = v;
      }
    }
  }
}

// ---------------------------------------------------------------------------
// Flash attention over the row-major-view batches: q,k,v = [64][1024][64] f16.
// 4 waves/block; each wave owns 16 query rows, full d=64 accumulator.
// S streamed in 64-wide chunks through LDS; online softmax in registers.
// K/V chunk s0+64 global loads overlap chunk s0 compute.
// ---------------------------------------------------------------------------
__global__ __launch_bounds__(128)
void mha_flash_kernel(const half_t* __restrict__ qh,
                      const half_t* __restrict__ kh,
                      const half_t* __restrict__ vh,
                      float* __restrict__ attn) {
  constexpr int LDT = 72;   // 64 + 8 pad halfs -> 144B row stride (16B mult)
  constexpr int LDP = 72;
  __shared__ __align__(16) half_t K_lds[64 * LDT];
  __shared__ __align__(16) half_t V_lds[64 * LDT];       // stored transposed [d][s]
  __shared__ __align__(16) half_t P_lds[4 * 16 * LDP];   // wave-private P tiles

  const int tid  = threadIdx.x;
  const int wid  = tid >> 5;
  const int lane = tid & 31;
  const int cmod = lane & 15;
  const int rhal = (lane >> 4) * 8;

  const int g    = blockIdx.y;                 // batch*head group (row-major view)
  const int row0 = blockIdx.x * 64 + wid * 16; // this wave's 16 query rows

  const size_t gbase = (size_t)g * SEQ * DHEAD;
  const half_t* qb = qh + gbase + (size_t)row0 * DHEAD;
  const half_t* kb = kh + gbase;
  const half_t* vb = vh + gbase;

  // Q fragments held for the whole kernel (d = 2 K-steps of 32)
  const v16h aq0 = load_frag_a(qb + 0, DHEAD);
  const v16h aq1 = load_frag_a(qb + 32, DHEAD);

  v8f zero = {};
  v8f O[4];                                    // 4 d-tiles of 16 cols, f32
#pragma unroll
  for (int t = 0; t < 4; ++t) O[t] = zero;
  float m_i[8], l_i[8];
#pragma unroll
  for (int r = 0; r < 8; ++r) { m_i[r] = -1e30f; l_i[r] = 0.f; }

  half_t* Pw = P_lds + wid * 16 * LDP;
  const float scale = 0.125f;                  // 1/sqrt(64)

  // Pipelined K/V chunk staging registers
  v8h kReg[4], vReg[4];
  auto load_chunk = [&](int s0) {
#pragma unroll
    for (int i = 0; i < 4; ++i) {
      const int seg = tid + i * 128;           // 512 segs of 8 halfs
      const int s   = seg >> 3;
      const int dd  = (seg & 7) * 8;
      kReg[i] = *(const v8h*)(kb + (size_t)(s0 + s) * DHEAD + dd);
      vReg[i] = *(const v8h*)(vb + (size_t)(s0 + s) * DHEAD + dd);
    }
  };
  auto store_chunk = [&]() {
#pragma unroll
    for (int i = 0; i < 4; ++i) {
      const int seg = tid + i * 128;
      const int s   = seg >> 3;
      const int dd  = (seg & 7) * 8;
      *(v8h*)&K_lds[s * LDT + dd] = kReg[i];
      const v8h vv = vReg[i];
#pragma unroll
      for (int j = 0; j < 8; ++j) V_lds[(dd + j) * LDT + s] = vv[j]; // transpose
    }
  };

  load_chunk(0);
  for (int s0 = 0; s0 < SEQ; s0 += 64) {
    __syncthreads();
    store_chunk();
    __syncthreads();
    if (s0 + 64 < SEQ) {
      load_chunk(s0 + 64);                     // overlap with compute below
      if (s0 + 128 < SEQ) {
        __builtin_prefetch(kb + (size_t)(s0 + 128 + (tid >> 3)) * DHEAD, 0, 3);
        __builtin_prefetch(vb + (size_t)(s0 + 128 + (tid >> 3)) * DHEAD, 0, 3);
      }
    }

    // scores = Q @ K^T : B-fragment column c = key (s0 + nt*16 + c), K-run = d
    v8f sc[4];
#pragma unroll
    for (int nt = 0; nt < 4; ++nt) {
      const v16h b0 = load_frag_b(&K_lds[(nt * 16) * LDT + 0], LDT);
      const v16h b1 = load_frag_b(&K_lds[(nt * 16) * LDT + 32], LDT);
      v8f z = zero;
      z = wmma_f16(aq0, b0, z);
      sc[nt] = wmma_f16(aq1, b1, z);
    }

    // Online softmax: row's 16 cols live in one VGPR across a 16-lane group
#pragma unroll
    for (int r = 0; r < 8; ++r) {
      float x[4];
#pragma unroll
      for (int nt = 0; nt < 4; ++nt) x[nt] = sc[nt][r] * scale;
      float mx = fmaxf(fmaxf(x[0], x[1]), fmaxf(x[2], x[3]));
#pragma unroll
      for (int off = 8; off > 0; off >>= 1)
        mx = fmaxf(mx, __shfl_xor(mx, off, 32));
      const float mnew = fmaxf(m_i[r], mx);
      const float corr = __expf(m_i[r] - mnew);
      float psum = 0.f;
      half_t ph[4];
#pragma unroll
      for (int nt = 0; nt < 4; ++nt) {
        const float p = __expf(x[nt] - mnew);
        psum += p;
        ph[nt] = (half_t)p;
      }
#pragma unroll
      for (int off = 8; off > 0; off >>= 1) psum += __shfl_xor(psum, off, 32);
      l_i[r] = l_i[r] * corr + psum;
      m_i[r] = mnew;
#pragma unroll
      for (int dt = 0; dt < 4; ++dt) O[dt][r] *= corr;
      const int prow = rhal + r;
#pragma unroll
      for (int nt = 0; nt < 4; ++nt)
        Pw[prow * LDP + nt * 16 + cmod] = ph[nt];
    }

    // O += P @ V : A = P (via wave-private LDS reshape), B = V^T rows
#pragma unroll
    for (int ks = 0; ks < 2; ++ks) {
      const v16h ap = load_frag_a(Pw + ks * 32, LDP);
#pragma unroll
      for (int dt = 0; dt < 4; ++dt) {
        const v16h bv = load_frag_b(&V_lds[(dt * 16) * LDT + ks * 32], LDT);
        O[dt] = wmma_f16(ap, bv, O[dt]);
      }
    }
  }

  // Normalize and store f32 attention output (flat row-major view buffer)
  const size_t obase = gbase + (size_t)row0 * DHEAD;
#pragma unroll
  for (int r = 0; r < 8; ++r) {
    const float inv = 1.f / l_i[r];
    const int row = rhal + r;
#pragma unroll
    for (int dt = 0; dt < 4; ++dt)
      attn[obase + (size_t)row * DHEAD + dt * 16 + cmod] = O[dt][r] * inv;
  }
}

// ---------------------------------------------------------------------------
extern "C" void kernel_launch(void* const* d_in, const int* in_sizes, int n_in,
                              void* d_out, int out_size, void* d_ws, size_t ws_size,
                              hipStream_t stream) {
  (void)in_sizes; (void)n_in; (void)out_size; (void)ws_size;
  const float* queries = (const float*)d_in[0];
  const float* keys    = (const float*)d_in[1];
  const float* values  = (const float*)d_in[2];
  const float* Wq = (const float*)d_in[3];
  const float* bq = (const float*)d_in[4];
  const float* Wk = (const float*)d_in[5];
  const float* bk = (const float*)d_in[6];
  const float* Wv = (const float*)d_in[7];
  const float* bv = (const float*)d_in[8];
  const float* Wo = (const float*)d_in[9];
  const float* bo = (const float*)d_in[10];
  float* out = (float*)d_out;

  // Workspace layout (~48 MB total)
  half_t* wqt = (half_t*)d_ws;
  half_t* wkt = wqt + (size_t)D_MODEL * D_MODEL;
  half_t* wvt = wkt + (size_t)D_MODEL * D_MODEL;
  half_t* wot = wvt + (size_t)D_MODEL * D_MODEL;
  half_t* qh  = wot + (size_t)D_MODEL * D_MODEL;
  half_t* kh  = qh + (size_t)MROWS * D_MODEL;
  half_t* vh  = kh + (size_t)MROWS * D_MODEL;
  float*  attn = (float*)(vh + (size_t)MROWS * D_MODEL);

  const dim3 tgrid(D_MODEL / 32, D_MODEL / 32), tblk(256);
  mha_transpose_cvt_kernel<<<tgrid, tblk, 0, stream>>>(Wq, wqt, D_MODEL, D_MODEL);
  mha_transpose_cvt_kernel<<<tgrid, tblk, 0, stream>>>(Wk, wkt, D_MODEL, D_MODEL);
  mha_transpose_cvt_kernel<<<tgrid, tblk, 0, stream>>>(Wv, wvt, D_MODEL, D_MODEL);
  mha_transpose_cvt_kernel<<<tgrid, tblk, 0, stream>>>(Wo, wot, D_MODEL, D_MODEL);

  const dim3 ggrid(D_MODEL / 128, MROWS / 128), gblk(256);
  mha_gemm_bias_kernel<true><<<ggrid, gblk, 0, stream>>>(
      queries, wqt, bq, qh, nullptr, MROWS, D_MODEL, D_MODEL);
  mha_gemm_bias_kernel<true><<<ggrid, gblk, 0, stream>>>(
      keys, wkt, bk, kh, nullptr, MROWS, D_MODEL, D_MODEL);
  mha_gemm_bias_kernel<true><<<ggrid, gblk, 0, stream>>>(
      values, wvt, bv, vh, nullptr, MROWS, D_MODEL, D_MODEL);

  const dim3 fgrid(SEQ / 64, GBATCH), fblk(128);
  mha_flash_kernel<<<fgrid, fblk, 0, stream>>>(qh, kh, vh, attn);

  mha_gemm_bias_kernel<false><<<ggrid, gblk, 0, stream>>>(
      attn, wot, bo, nullptr, out, MROWS, D_MODEL, D_MODEL);
}
